// ModulatedDeformConvPack_62886911148488
// MI455X (gfx1250) — matrix operand order repaired and checked by reading, not compiled, not run on hardware
//
#include <hip/hip_runtime.h>

typedef __attribute__((ext_vector_type(16))) __bf16 v16bf;
typedef __attribute__((ext_vector_type(8)))  float  v8f;

#define Bn 8
#define Cn 64
#define Hn 96
#define Wn 96
#define HW 9216            // 96*96
#define Kn 9               // 3x3
#define OCn 64
#define KDIM 576           // Cn*Kn
#define Mtot 73728         // Bn*HW
#define BPAD 584           // 576 + 8 bf16 pad for LDS B rows

// fp32 -> bf16 (round-to-nearest-even), pure integer ops
__device__ inline unsigned short f2bfu(float f) {
    unsigned u = __float_as_uint(f);
    u += 0x7FFFu + ((u >> 16) & 1u);
    return (unsigned short)(u >> 16);
}

// ---------------- Stage 0: weight fp32 -> bf16, transpose to [oc][kk*64+c] ----
__global__ __launch_bounds__(256) void k_weights(const float* __restrict__ w,
                                                 unsigned short* __restrict__ wB) {
    int t = blockIdx.x * 256 + threadIdx.x;       // 64*576 = 36864 threads
    int oc  = t / KDIM;
    int rem = t - oc * KDIM;
    int c   = rem / Kn;
    int kk  = rem - c * Kn;
    wB[oc * KDIM + kk * 64 + c] = f2bfu(w[t]);
}

// ---------------- Stage 1: offset conv + sigmoid + corner precompute ---------
__global__ __launch_bounds__(256) void k_offsets(const float* __restrict__ x,
                                                 const float* __restrict__ w_off,
                                                 const float* __restrict__ b_off,
                                                 uint2* __restrict__ idxA,
                                                 float4* __restrict__ wgtA) {
    __shared__ float sw[27 * KDIM];               // 62208 B
    for (int i = threadIdx.x; i < 27 * KDIM; i += 256) sw[i] = w_off[i];
    __syncthreads();

    int t   = blockIdx.x * 256 + threadIdx.x;     // 73728 threads
    int b   = t / HW;
    int pos = t - b * HW;
    int ho  = pos / Wn;
    int wo  = pos - ho * Wn;

    float acc[27];
#pragma unroll
    for (int ch = 0; ch < 27; ++ch) acc[ch] = 0.f;

    const float* xb = x + (size_t)b * Cn * HW;
    for (int c = 0; c < Cn; ++c) {
        const float* xc = xb + c * HW;
#pragma unroll
        for (int dy = 0; dy < 3; ++dy) {
            int y = ho + dy - 1;
            bool yin = (y >= 0) && (y < Hn);
#pragma unroll
            for (int dx = 0; dx < 3; ++dx) {
                int xx = wo + dx - 1;
                float xv = (yin && xx >= 0 && xx < Wn) ? xc[y * Wn + xx] : 0.f;
                int tap = c * 9 + dy * 3 + dx;
#pragma unroll
                for (int ch = 0; ch < 27; ++ch)
                    acc[ch] = fmaf(xv, sw[ch * KDIM + tap], acc[ch]);
            }
        }
    }

#pragma unroll
    for (int k = 0; k < Kn; ++k) {
        float py = acc[k]      + b_off[k]      + (float)(ho + (k / 3) - 1);
        float px = acc[9 + k]  + b_off[9 + k]  + (float)(wo + (k % 3) - 1);
        float ml = acc[18 + k] + b_off[18 + k];
        float mask = 1.0f / (1.0f + expf(-ml));
        float y0f = floorf(py), x0f = floorf(px);
        float ly = py - y0f, lx = px - x0f;
        int y0 = (int)y0f, x0 = (int)x0f;
        unsigned idc[4]; float wts[4];
#pragma unroll
        for (int q = 0; q < 4; ++q) {
            int dy = q >> 1, dx = q & 1;
            int yy = y0 + dy, xx2 = x0 + dx;
            bool v = (yy >= 0) && (yy < Hn) && (xx2 >= 0) && (xx2 < Wn);
            int yi = min(max(yy, 0), Hn - 1);
            int xi = min(max(xx2, 0), Wn - 1);
            idc[q] = (unsigned)(yi * Wn + xi);
            float wy = dy ? ly : (1.f - ly);
            float wx = dx ? lx : (1.f - lx);
            wts[q] = v ? (wy * wx * mask) : 0.f;
        }
        size_t e = (size_t)(b * Kn + k) * HW + pos;
        idxA[e] = make_uint2(idc[0] | (idc[1] << 16), idc[2] | (idc[3] << 16));
        wgtA[e] = make_float4(wts[0], wts[1], wts[2], wts[3]);
    }
}

// ---------------- Stage 2: fused im2col-sample + bf16 WMMA implicit GEMM -----
// Block: 128 threads (4 waves). Each block: 64 M-rows x 64 OC. K loop: 18 x 32.
__global__ __launch_bounds__(128) void k_gemm(const float* __restrict__ x,
                                              const unsigned short* __restrict__ wB,
                                              const uint2* __restrict__ idxA,
                                              const float4* __restrict__ wgtA,
                                              const float* __restrict__ bias,
                                              float* __restrict__ out) {
    __shared__ __align__(16) unsigned short Blds[OCn * BPAD];   // 74752 B
    {   // stage full bf16 B into LDS (dword copies, padded rows)
        const unsigned* src = (const unsigned*)wB;              // 18432 dwords
        unsigned* dst = (unsigned*)Blds;
        for (int i = threadIdx.x; i < OCn * (KDIM / 2); i += 128) {
            int r  = i / (KDIM / 2);
            int c2 = i - r * (KDIM / 2);
            dst[r * (BPAD / 2) + c2] = src[i];
        }
    }
    __syncthreads();

    const int wave   = threadIdx.x >> 5;
    const int lane   = threadIdx.x & 31;
    const int laneHi = lane >> 4;
    const int lane15 = lane & 15;
    const int mBase  = blockIdx.x * 64;
    const int row    = mBase + wave * 16 + lane15;   // A-fragment row
    const int b      = row / HW;                     // tile never crosses batch
    const int pos    = row - b * HW;
    const float* xb  = x + (size_t)b * Cn * HW;
    const size_t pb  = (size_t)b * Kn * HW + pos;

    v8f acc[4];
    v8f zero = {0.f, 0.f, 0.f, 0.f, 0.f, 0.f, 0.f, 0.f};
#pragma unroll
    for (int nt = 0; nt < 4; ++nt) acc[nt] = zero;

#pragma unroll 1
    for (int chunk = 0; chunk < KDIM / 32; ++chunk) {
        const int kc0 = chunk * 32;
        const int kk  = kc0 >> 6;        // k-major reduction: kk constant per chunk
        const int c0  = kc0 & 63;
        const uint2  id = idxA[pb + (size_t)kk * HW];
        const float4 w4 = wgtA[pb + (size_t)kk * HW];
        const int i00 = id.x & 0xFFFF, i01 = id.x >> 16;
        const int i10 = id.y & 0xFFFF, i11 = id.y >> 16;

        union { v16bf v; unsigned short s[16]; } ua;
#pragma unroll
        for (int e = 0; e < 16; ++e) {
            // A 16x32 layout (contiguous, matches B): K = e + 16*laneHi
            const int K = e + 16 * laneHi;
            const float* xc = xb + (size_t)(c0 + K) * HW;
            float v = w4.x * xc[i00] + w4.y * xc[i01]
                    + w4.z * xc[i10] + w4.w * xc[i11];
            ua.s[e] = f2bfu(v);
        }

#pragma unroll
        for (int nt = 0; nt < 4; ++nt) {
            const int n = nt * 16 + lane15;          // B layout: N=lane, K contiguous
            union { v16bf v; uint4 q[2]; } ub;
            const uint4* bp = (const uint4*)&Blds[n * BPAD + kc0 + laneHi * 16];
            ub.q[0] = bp[0];
            ub.q[1] = bp[1];
            acc[nt] = __builtin_amdgcn_wmma_f32_16x16x32_bf16(
                false, ua.v, false, ub.v, (short)0, acc[nt], false, false);
        }
    }

    // epilogue: C layout — VGPR g holds M = g + 8*laneHi, N = lane15
#pragma unroll
    for (int nt = 0; nt < 4; ++nt) {
        const int oc = nt * 16 + lane15;
        const float bv = bias[oc];
#pragma unroll
        for (int g = 0; g < 8; ++g) {
            const int orow = mBase + wave * 16 + g + 8 * laneHi;
            const int opos = orow - b * HW;
            out[((size_t)b * OCn + oc) * HW + opos] = acc[nt][g] + bv;
        }
    }
}

extern "C" void kernel_launch(void* const* d_in, const int* in_sizes, int n_in,
                              void* d_out, int out_size, void* d_ws, size_t ws_size,
                              hipStream_t stream) {
    const float* x      = (const float*)d_in[0];
    const float* w_off  = (const float*)d_in[1];
    const float* b_off  = (const float*)d_in[2];
    const float* weight = (const float*)d_in[3];
    const float* bias   = (const float*)d_in[4];

    char* ws = (char*)d_ws;
    // workspace layout (16-byte aligned sections)
    const size_t OFF_WB  = 0;                               // 36864 * 2   = 73728 B
    const size_t OFF_IDX = 73728;                           // 663552 * 8  = 5308416 B
    const size_t OFF_WGT = 73728 + 5308416;                 // 663552 * 16 = 10616832 B
    unsigned short* wB   = (unsigned short*)(ws + OFF_WB);
    uint2*          idxA = (uint2*)(ws + OFF_IDX);
    float4*         wgtA = (float4*)(ws + OFF_WGT);

    // Stage 0: 64*576 weight elements
    k_weights<<<(OCn * KDIM) / 256, 256, 0, stream>>>(weight, wB);
    // Stage 1: one thread per (b, pos)
    k_offsets<<<Mtot / 256, 256, 0, stream>>>(x, w_off, b_off, idxA, wgtA);
    // Stage 2: 64-row M tiles
    k_gemm<<<Mtot / 64, 128, 0, stream>>>(x, wB, idxA, wgtA, bias, (float*)d_out);
}
